// point_transformer_76158360093246
// MI455X (gfx1250) — matrix operand order repaired and checked by reading, not compiled, not run on hardware
//
#include <hip/hip_runtime.h>
#include <cstddef>
#include <cstdint>

typedef __attribute__((ext_vector_type(16))) _Float16 v16h;
typedef __attribute__((ext_vector_type(8)))  float    v8f;

#define DEV __device__ __forceinline__

static constexpr int B_   = 2;
static constexpr int N_   = 4096;
static constexpr int DIM_ = 32;
static constexpr int K_   = 16;

// ---------------- WMMA helpers (CDNA5 16x16x32 f16 -> f32) ----------------

DEV v8f wmma_f32_f16(v16h a, v16h b, v8f c) {
  // (neg_a, A, neg_b, B, c_mod, C, reuse_a, reuse_b)
  return __builtin_amdgcn_wmma_f32_16x16x32_f16(false, a, false, b, (short)0, c,
                                                false, false);
}

// A tile: 16 rows x kCols (<=32) from row-major f32 (leading dim ld), packed to
// the CDNA5 16-bit A layout: lanes 0-15 = rows, VGPR0-3 hold K={0..7}+8*half,
// VGPR4-7 hold K={16..23}+8*half.
DEV v16h load_a_rm(const float* base, int ld, int kCols, int lane) {
  const int m = lane & 15, half = lane >> 4;
  v16h a;
  #pragma unroll
  for (int r = 0; r < 8; ++r) {
    const int k0 = (r < 4) ? (2*r + 8*half) : (16 + 2*(r - 4) + 8*half);
    const float x0 = (k0     < kCols) ? base[(size_t)m*ld + k0    ] : 0.0f;
    const float x1 = (k0 + 1 < kCols) ? base[(size_t)m*ld + k0 + 1] : 0.0f;
    a[2*r    ] = (_Float16)x0;
    a[2*r + 1] = (_Float16)x1;
  }
  return a;
}

// B tile: kRows (<=32) x 16 columns (columns 16*ntile..16*ntile+15) from
// row-major W[fan_in][ldN]. Same K order as A: elems 0-7 K={0..7}+8*half, elems 8-15 K={16..23}+8*half.
DEV v16h load_b_rm(const float* W, int ldN, int kRows, int ntile, int lane) {
  const int n = (lane & 15) + ntile*16, khalf = lane >> 4;
  v16h b;
  #pragma unroll
  for (int r = 0; r < 8; ++r) {
    const int k = (r < 4) ? (2*r + 8*khalf) : (16 + 2*(r - 4) + 8*khalf);
    const float x0 = (k     < kRows) ? W[(size_t)k*ldN + n]       : 0.0f;
    const float x1 = (k + 1 < kRows) ? W[(size_t)(k + 1)*ldN + n] : 0.0f;
    b[2*r    ] = (_Float16)x0;
    b[2*r + 1] = (_Float16)x1;
  }
  return b;
}

// Bias + optional ReLU on a C/D accumulator tile (column = (lane&15)+16*ntile,
// rows = (lane>>4)*8 + r).
DEV void bias_act(v8f& d, const float* bias, int ntile, int lane, bool relu) {
  const int col = (lane & 15) + ntile*16;
  const float bb = bias[col];
  #pragma unroll
  for (int r = 0; r < 8; ++r) {
    float x = d[r] + bb;
    d[r] = relu ? fmaxf(x, 0.0f) : x;
  }
}

// Scatter a D tile to row-major memory (LDS or global), leading dim ld.
DEV void store_tile(float* base, size_t ld, v8f d, int ntile, int lane) {
  const int col = (lane & 15) + ntile*16, mh = (lane >> 4)*8;
  #pragma unroll
  for (int r = 0; r < 8; ++r) base[(size_t)(mh + r)*ld + col] = d[r];
}

// ---------------- Kernel 1: f = relu(x@W1), q/k/v = relu(f@W*) ----------------

__global__ __launch_bounds__(256) void pt_qkv_kernel(
    const float* __restrict__ feat,
    const float* __restrict__ W1, const float* __restrict__ b1,
    const float* __restrict__ Wq, const float* __restrict__ bq,
    const float* __restrict__ Wk, const float* __restrict__ bk,
    const float* __restrict__ Wv, const float* __restrict__ bv,
    float* __restrict__ qo, float* __restrict__ ko, float* __restrict__ vo)
{
  __shared__ float S[8][16*32];                 // per-wave staging tile
  const int lane = threadIdx.x & 31, wave = threadIdx.x >> 5;
  const int pt0 = (blockIdx.x*8 + wave) * 16;   // 16 points per wave
  float* St = S[wave];
  v8f zero = {};

  // f = relu(feature @ W1 + b1)
  v16h Af = load_a_rm(feat + (size_t)pt0*DIM_, DIM_, 32, lane);
  v8f f0 = wmma_f32_f16(Af, load_b_rm(W1, 32, 32, 0, lane), zero);
  v8f f1 = wmma_f32_f16(Af, load_b_rm(W1, 32, 32, 1, lane), zero);
  bias_act(f0, b1, 0, lane, true);
  bias_act(f1, b1, 1, lane, true);
  store_tile(St, 32, f0, 0, lane);              // D-layout -> row-major in LDS
  store_tile(St, 32, f1, 1, lane);
  v16h Afx = load_a_rm(St, 32, 32, lane);       // reload as A operand (f16)

  // q = relu(f @ Wq + bq)
  v8f d0 = wmma_f32_f16(Afx, load_b_rm(Wq, 32, 32, 0, lane), zero);
  v8f d1 = wmma_f32_f16(Afx, load_b_rm(Wq, 32, 32, 1, lane), zero);
  bias_act(d0, bq, 0, lane, true); bias_act(d1, bq, 1, lane, true);
  store_tile(qo + (size_t)pt0*DIM_, DIM_, d0, 0, lane);
  store_tile(qo + (size_t)pt0*DIM_, DIM_, d1, 1, lane);

  // k = relu(f @ Wk + bk)
  d0 = wmma_f32_f16(Afx, load_b_rm(Wk, 32, 32, 0, lane), zero);
  d1 = wmma_f32_f16(Afx, load_b_rm(Wk, 32, 32, 1, lane), zero);
  bias_act(d0, bk, 0, lane, true); bias_act(d1, bk, 1, lane, true);
  store_tile(ko + (size_t)pt0*DIM_, DIM_, d0, 0, lane);
  store_tile(ko + (size_t)pt0*DIM_, DIM_, d1, 1, lane);

  // v = relu(f @ Wv + bv)
  d0 = wmma_f32_f16(Afx, load_b_rm(Wv, 32, 32, 0, lane), zero);
  d1 = wmma_f32_f16(Afx, load_b_rm(Wv, 32, 32, 1, lane), zero);
  bias_act(d0, bv, 0, lane, true); bias_act(d1, bv, 1, lane, true);
  store_tile(vo + (size_t)pt0*DIM_, DIM_, d0, 0, lane);
  store_tile(vo + (size_t)pt0*DIM_, DIM_, d1, 1, lane);
}

// ---------------- Kernel 2: brute-force kNN (top-16 by squared distance) ------
// One wave per query point. Keys are 32-bit: dist2 float bits with the low 12
// mantissa bits replaced by the candidate index (N=4096 -> 12 bits). Monotone
// in distance; u32 min/max compare-exchange is 2 VALU ops per slot.

__global__ __launch_bounds__(256) void pt_knn_kernel(const float* __restrict__ pos,
                                                     int* __restrict__ knn)
{
  const int lane = threadIdx.x & 31, wave = threadIdx.x >> 5;
  const int pt = blockIdx.x*8 + wave;
  const int b = pt / N_, i = pt - b*N_;
  const float* pb = pos + (size_t)b*N_*3;
  const float3* p3 = (const float3*)pb;         // 12B packed -> global_load_b96
  const float3 pi = p3[i];

  unsigned best[16];                            // per-lane ascending top-16
  #pragma unroll
  for (int s = 0; s < 16; ++s) best[s] = 0xFFFFFFFFu;

  for (int c = 0; c < N_/32; ++c) {
    const int j = c*32 + lane;
    // prefetch two chunks ahead (pos is L2/WGP$-resident; hides latency)
    if (c + 2 < N_/32) __builtin_prefetch(&p3[j + 64], 0, 1);
    const float3 pj = p3[j];
    const float dx = pj.x - pi.x, dy = pj.y - pi.y, dz = pj.z - pi.z;
    const float d2 = dx*dx + dy*dy + dz*dz;
    unsigned key = (__float_as_uint(d2) & ~0xFFFu) | (unsigned)j;
    // wave-uniform skip: enter the chain only if some lane improves its list
    if (__any(key < best[15])) {
      #pragma unroll
      for (int s = 0; s < 16; ++s) {            // v_min_u32 / v_max_u32 chain
        const unsigned lo = key < best[s] ? key : best[s];
        const unsigned hi = key < best[s] ? best[s] : key;
        best[s] = lo;
        key = hi;
      }
    }
  }

  // merge 32 sorted lists: 16 rounds of wave-min + pop (wave32 shuffles)
  for (int t = 0; t < K_; ++t) {
    unsigned g = best[0];
    #pragma unroll
    for (int off = 16; off >= 1; off >>= 1) {
      const unsigned o = __shfl_xor(g, off, 32);
      g = (o < g) ? o : g;
    }
    const unsigned long long bal = __ballot(best[0] == g);
    const int fl = __ffsll((unsigned long long)bal) - 1;
    if (lane == fl) {                           // pop winner's head
      #pragma unroll
      for (int s = 0; s < 15; ++s) best[s] = best[s + 1];
      best[15] = 0xFFFFFFFFu;
    }
    if (lane == 0) knn[(size_t)pt*K_ + t] = (int)(g & 0xFFFu);
  }
}

// ---------------- Kernel 3: fused neighborhood attention ----------------------

__global__ __launch_bounds__(256) void pt_attn_kernel(
    const float* __restrict__ pos,
    const float* __restrict__ qg, const float* __restrict__ kg,
    const float* __restrict__ vg, const int* __restrict__ knn,
    const float* __restrict__ Wp1, const float* __restrict__ bp1,
    const float* __restrict__ Wp2, const float* __restrict__ bp2,
    const float* __restrict__ Wa1, const float* __restrict__ ba1,
    const float* __restrict__ Wa2, const float* __restrict__ ba2,
    const float* __restrict__ W2,  const float* __restrict__ b2,
    float* __restrict__ out)
{
  __shared__ float S[8][16*32];
  __shared__ float OV[8][32];
  const int lane = threadIdx.x & 31, wave = threadIdx.x >> 5;
  const int pt = blockIdx.x*8 + wave;           // one wave per point
  const int b = pt / N_, i = pt - b*N_;
  const int* nb = knn + (size_t)pt*K_;
  const float* pb = pos + (size_t)b*N_*3;
  float* St = S[wave];
  v8f zero = {};
  const int n = lane & 15, half = lane >> 4, mh = half*8;

  // pos-MLP hidden layer (3->32), VALU: lane handles neighbor (lane&15), 16 cols
  {
    const int m = lane & 15;
    const int j = nb[m];
    const float rx = pb[(size_t)j*3 + 0] - pb[(size_t)i*3 + 0];
    const float ry = pb[(size_t)j*3 + 1] - pb[(size_t)i*3 + 1];
    const float rz = pb[(size_t)j*3 + 2] - pb[(size_t)i*3 + 2];
    const int c0 = half*16;
    #pragma unroll
    for (int c = 0; c < 16; ++c) {
      const int col = c0 + c;
      const float h = rx*Wp1[col] + ry*Wp1[32 + col] + rz*Wp1[64 + col] + bp1[col];
      St[m*32 + col] = fmaxf(h, 0.0f);
    }
  }

  // pos_emb = relu(h @ Wp2 + bp2) via WMMA (16 neighbors x 32)
  v16h Ah = load_a_rm(St, 32, 32, lane);
  v8f pe0 = wmma_f32_f16(Ah, load_b_rm(Wp2, 32, 32, 0, lane), zero);
  v8f pe1 = wmma_f32_f16(Ah, load_b_rm(Wp2, 32, 32, 1, lane), zero);
  bias_act(pe0, bp2, 0, lane, true);
  bias_act(pe1, bp2, 1, lane, true);

  int jn[8];
  #pragma unroll
  for (int r = 0; r < 8; ++r) jn[r] = nb[mh + r];

  // s = (q[gather] - k[i]) + pos_emb  -> staged row-major in LDS
  #pragma unroll
  for (int t = 0; t < 2; ++t) {
    const int col = t*16 + n;
    const float kv = kg[(size_t)pt*DIM_ + col];
    #pragma unroll
    for (int r = 0; r < 8; ++r) {
      const float qv = qg[((size_t)b*N_ + jn[r])*DIM_ + col];
      const float pe = t ? pe1[r] : pe0[r];
      St[(size_t)(mh + r)*32 + col] = qv - kv + pe;
    }
  }

  // a = relu(s @ Wa1 + ba1)  (32 -> 16)
  v16h As = load_a_rm(St, 32, 32, lane);
  v8f a0 = wmma_f32_f16(As, load_b_rm(Wa1, 16, 32, 0, lane), zero);
  bias_act(a0, ba1, 0, lane, true);
  store_tile(St, 32, a0, 0, lane);

  // logits = a @ Wa2 + ba2   (16 -> 32, K zero-padded to 32)
  v16h Aa = load_a_rm(St, 32, 16, lane);
  v8f l0 = wmma_f32_f16(Aa, load_b_rm(Wa2, 32, 16, 0, lane), zero);
  v8f l1 = wmma_f32_f16(Aa, load_b_rm(Wa2, 32, 16, 1, lane), zero);
  bias_act(l0, ba2, 0, lane, false);
  bias_act(l1, ba2, 1, lane, false);

  // softmax over the 16 neighbors per channel; column lives in lane pair
  // (l, l+16) -> single __shfl_xor(.,16) for max/sum/accumulate.
  #pragma unroll
  for (int t = 0; t < 2; ++t) {
    v8f L = t ? l1 : l0;
    float mx = L[0];
    #pragma unroll
    for (int r = 1; r < 8; ++r) mx = fmaxf(mx, L[r]);
    mx = fmaxf(mx, __shfl_xor(mx, 16, 32));
    float e[8], sm = 0.0f;
    #pragma unroll
    for (int r = 0; r < 8; ++r) { e[r] = __expf(L[r] - mx); sm += e[r]; }
    sm += __shfl_xor(sm, 16, 32);
    const float inv = 1.0f / sm;
    const int col = t*16 + n;
    float acc = 0.0f;
    #pragma unroll
    for (int r = 0; r < 8; ++r)
      acc += vg[((size_t)b*N_ + jn[r])*DIM_ + col] * e[r];
    acc += __shfl_xor(acc, 16, 32);
    if (half == 0) OV[wave][col] = acc * inv;
  }

  // out = ov @ W2 + b2 (no relu); lane = output channel
  float r2 = b2[lane];
  #pragma unroll
  for (int d = 0; d < 32; ++d) r2 += OV[wave][d] * W2[(size_t)d*32 + lane];
  out[(size_t)pt*DIM_ + lane] = r2;
}

// ---------------- Host launch --------------------------------------------------

extern "C" void kernel_launch(void* const* d_in, const int* in_sizes, int n_in,
                              void* d_out, int out_size, void* d_ws, size_t ws_size,
                              hipStream_t stream) {
  (void)in_sizes; (void)n_in; (void)out_size; (void)ws_size;
  const float* feature = (const float*)d_in[0];
  const float* pos     = (const float*)d_in[1];
  const float* W1  = (const float*)d_in[2];  const float* b1  = (const float*)d_in[3];
  const float* Wq  = (const float*)d_in[4];  const float* bq  = (const float*)d_in[5];
  const float* Wk  = (const float*)d_in[6];  const float* bk  = (const float*)d_in[7];
  const float* Wv  = (const float*)d_in[8];  const float* bv  = (const float*)d_in[9];
  const float* Wp1 = (const float*)d_in[10]; const float* bp1 = (const float*)d_in[11];
  const float* Wp2 = (const float*)d_in[12]; const float* bp2 = (const float*)d_in[13];
  const float* Wa1 = (const float*)d_in[14]; const float* ba1 = (const float*)d_in[15];
  const float* Wa2 = (const float*)d_in[16]; const float* ba2 = (const float*)d_in[17];
  const float* W2  = (const float*)d_in[18]; const float* b2  = (const float*)d_in[19];
  float* out = (float*)d_out;

  const size_t npts = (size_t)B_ * N_;          // 8192
  float* q   = (float*)d_ws;
  float* k   = q + npts*DIM_;
  float* v   = k + npts*DIM_;
  int*   knn = (int*)(v + npts*DIM_);

  pt_qkv_kernel<<<(int)(npts/(16*8)), 256, 0, stream>>>(
      feature, W1, b1, Wq, bq, Wk, bk, Wv, bv, q, k, v);
  pt_knn_kernel<<<(int)(npts/8), 256, 0, stream>>>(pos, knn);
  pt_attn_kernel<<<(int)(npts/8), 256, 0, stream>>>(
      pos, q, k, v, knn, Wp1, bp1, Wp2, bp2, Wa1, ba1, Wa2, ba2, W2, b2, out);
}
